// Attention_781684048063
// MI455X (gfx1250) — matrix operand order repaired and checked by reading, not compiled, not run on hardware
//
#include <hip/hip_runtime.h>

// Problem constants (B=1, H=W=64, C=768, nh=12, hd=64)
#define NTOK  4096      // H*W
#define CDIM  768
#define QKVN  2304      // 3*C
#define NHEAD 12
#define HD    64

typedef __attribute__((ext_vector_type(16))) __bf16 v16bf;
typedef __attribute__((ext_vector_type(8)))  __bf16 v8bf;
typedef __attribute__((ext_vector_type(8)))  float  v8f;

// ---------- bf16 helpers (bit-level, round-to-nearest-even) ----------
__device__ __forceinline__ __bf16 f2bf(float f) {
  unsigned int u = __builtin_bit_cast(unsigned int, f);
  unsigned int r = (u + 0x7FFFu + ((u >> 16) & 1u)) >> 16;
  unsigned short s = (unsigned short)r;
  return __builtin_bit_cast(__bf16, s);
}
__device__ __forceinline__ float bf2f(__bf16 b) {
  unsigned int u = ((unsigned int)__builtin_bit_cast(unsigned short, b)) << 16;
  return __builtin_bit_cast(float, u);
}
// Pack two 8-element contiguous bf16 groups into a 16-element WMMA fragment
__device__ __forceinline__ v16bf pack16(const __bf16* p0, const __bf16* p1) {
  v8bf lo = *(const v8bf*)p0;
  v8bf hi = *(const v8bf*)p1;
  v16bf r;
#pragma unroll
  for (int j = 0; j < 8; ++j) { r[j] = lo[j]; r[j + 8] = hi[j]; }
  return r;
}

// ---------- elementwise fp32 -> bf16 ----------
__global__ void f32_to_bf16_kernel(const float* __restrict__ in, __bf16* __restrict__ out, int n) {
  int i = blockIdx.x * blockDim.x + threadIdx.x;
  if (i < n) out[i] = f2bf(in[i]);
}

// ---------- transpose + convert: in (K x N fp32, row-major) -> out (N x K bf16) ----------
__global__ void transpose_f32_to_bf16_kernel(const float* __restrict__ in, __bf16* __restrict__ out,
                                             int K, int N) {
  int i = blockIdx.x * blockDim.x + threadIdx.x;
  if (i >= K * N) return;
  int k = i % K;
  int n = i / K;
  out[(size_t)n * K + k] = f2bf(in[(size_t)k * N + n]);
}

// ---------- V transpose: vT[(head*64+d)*4096 + n] = qkv[n*2304 + 1536 + head*64 + d] ----------
__global__ void transpose_v_kernel(const __bf16* __restrict__ qkv, __bf16* __restrict__ vT) {
  int i = blockIdx.x * blockDim.x + threadIdx.x;
  if (i >= NHEAD * HD * NTOK) return;
  int n   = i & (NTOK - 1);
  int hd_ = i >> 12;             // head*64 + d  (0..767)
  vT[i] = qkv[(size_t)n * QKVN + 2 * CDIM + hd_];
}

// ---------- decomposed rel-pos bias tables: out[head][m][k2] = dot(q[head,m,:], tab[idx,:]) ----------
__global__ void rel_bias_kernel(const __bf16* __restrict__ qkv, const float* __restrict__ tab,
                                float* __restrict__ out, int isW) {
  int i = blockIdx.x * blockDim.x + threadIdx.x;
  if (i >= NHEAD * NTOK * HD) return;
  int k2   = i & 63;
  int m    = (i >> 6) & (NTOK - 1);
  int head = i >> 18;
  int pos  = isW ? (m & 63) : (m >> 6);
  int idx  = pos - k2 + 63;
  const __bf16* qp = qkv + (size_t)m * QKVN + head * HD;  // q slice (bias included)
  const float*  tp = tab + (size_t)idx * HD;
  float acc = 0.f;
#pragma unroll 8
  for (int d = 0; d < HD; ++d) acc += bf2f(qp[d]) * tp[d];
  out[i] = acc;
}

// ---------- WMMA GEMM: C = A(MxK bf16) @ Bt^T (Bt is NxK bf16) + bias ----------
// One wave computes a 16x64 output tile (4 accumulators), K-chunks of 32.
template <bool OUT_F32>
__global__ void gemm_bf16_kernel(const __bf16* __restrict__ A, const __bf16* __restrict__ Bt,
                                 const float* __restrict__ bias,
                                 __bf16* __restrict__ outb, float* __restrict__ outf,
                                 int M, int N, int K) {
  const int lane = threadIdx.x & 31;
  const int wid  = threadIdx.x >> 5;
  const int nsupers = N >> 6;
  const int gw = blockIdx.x * (blockDim.x >> 5) + wid;
  if (gw >= (M >> 4) * nsupers) return;
  const int mtile = gw / nsupers;
  const int nsup  = gw % nsupers;
  const int half  = lane >> 4;
  const int lm    = lane & 15;

  v8f c[4];
#pragma unroll
  for (int t = 0; t < 4; ++t) c[t] = (v8f){0.f,0.f,0.f,0.f,0.f,0.f,0.f,0.f};

  const __bf16* arow = A + (size_t)(mtile * 16 + lm) * K;
  for (int kc = 0; kc < K; kc += 32) {
    // A fragment (16x32): lane holds row lm; K groups half*8+[0..7] and 16+half*8+[0..7]
    const __bf16* ap = arow + kc + half * 8;
    v16bf a = pack16(ap, ap + 16);
#pragma unroll
    for (int t = 0; t < 4; ++t) {
      // B fragment (32x16): lane holds column; K groups half*8+[0..7] and 16+half*8+[0..7] in Bt
      const int col = nsup * 64 + t * 16 + lm;
      const __bf16* bp = Bt + (size_t)col * K + kc + half * 8;
      v16bf b = pack16(bp, bp + 16);
      c[t] = __builtin_amdgcn_wmma_f32_16x16x32_bf16(false, a, false, b, (short)0, c[t],
                                                     false, false);
    }
  }
#pragma unroll
  for (int t = 0; t < 4; ++t) {
    const int col = nsup * 64 + t * 16 + lm;
    const float bv = bias ? bias[col] : 0.f;
#pragma unroll
    for (int r = 0; r < 8; ++r) {
      const int mr = mtile * 16 + r + half * 8;   // C/D layout: vgpr r, lane-half -> row
      float v = c[t][r] + bv;
      if (OUT_F32) outf[(size_t)mr * N + col] = v;
      else         outb[(size_t)mr * N + col] = f2bf(v);
    }
  }
}

// ---------- flash attention: one wave per (head, 16-query tile) ----------
// Outer loop processes 64 keys (two 32-key blocks with compile-time parity) so
// all rel_w bias values live in registers and rel_h is loaded once per 64 keys.
__global__ void __launch_bounds__(256)
flash_attn_kernel(const __bf16* __restrict__ qkv,   // 4096 x 2304 (q|k|v per head)
                  const __bf16* __restrict__ vT,    // (12*64) x 4096
                  const float* __restrict__ relh,   // [head][m][64]
                  const float* __restrict__ relw,   // [head][m][64]
                  __bf16* __restrict__ outb)        // 4096 x 768
{
  __shared__ __align__(16) __bf16 pbuf[8][16 * 32];   // per-wave P tile (16 x 32)
  const int lane = threadIdx.x & 31;
  const int wid  = threadIdx.x >> 5;
  const int gw   = blockIdx.x * 8 + wid;
  if (gw >= NHEAD * (NTOK / 16)) return;
  const int head  = gw >> 8;          // 256 query-tiles per head
  const int qtile = gw & 255;
  const int half  = lane >> 4;
  const int lm    = lane & 15;
  const int m0    = qtile * 16;
  const float scale = 0.125f;         // hd^-0.5

  // Q fragments for the whole tile: two 16x32 chunks over d
  v16bf aq[2];
  {
    const __bf16* qp = qkv + (size_t)(m0 + lm) * QKVN + head * HD;
#pragma unroll
    for (int kk = 0; kk < 2; ++kk) {
      const __bf16* p = qp + kk * 32 + half * 8;
      aq[kk] = pack16(p, p + 16);
    }
  }

  float mrow[8], lrow[8];
  v8f o[4];
#pragma unroll
  for (int r = 0; r < 8; ++r) { mrow[r] = -1e30f; lrow[r] = 0.f; }
#pragma unroll
  for (int t = 0; t < 4; ++t) o[t] = (v8f){0.f,0.f,0.f,0.f,0.f,0.f,0.f,0.f};

  const float* rh = relh + ((size_t)head * NTOK + m0) * HD;
  const float* rw = relw + ((size_t)head * NTOK + m0) * HD;

  // rel_w bias: kw = (kb*32 + t*16 + lm) & 63 = (kb&1)*32 + t*16 + lm -> only
  // 2 (parity) x 2 (t) values per row r. Preload all 32 into registers.
  float bwreg[2][2][8];
#pragma unroll
  for (int p = 0; p < 2; ++p)
#pragma unroll
    for (int t = 0; t < 2; ++t)
#pragma unroll
      for (int r = 0; r < 8; ++r)
        bwreg[p][t][r] = rw[(r + half * 8) * HD + p * 32 + t * 16 + lm];

  for (int kbb = 0; kbb < NTOK / 64; ++kbb) {
    // rel_h bias: kh = key/64 = kbb, constant for these 64 keys
    float bh[8];
#pragma unroll
    for (int r = 0; r < 8; ++r) bh[r] = rh[(r + half * 8) * HD + kbb];

#pragma unroll
    for (int p = 0; p < 2; ++p) {           // two 32-key blocks, compile-time parity
      const int k0 = kbb * 64 + p * 32;
      // ---- S = Q K^T (16 x 32) as two 16x16 C fragments ----
      v8f s[2];
#pragma unroll
      for (int t = 0; t < 2; ++t) {
        s[t] = (v8f){0.f,0.f,0.f,0.f,0.f,0.f,0.f,0.f};
        const int key = k0 + t * 16 + lm;
        const __bf16* kp = qkv + (size_t)key * QKVN + CDIM + head * HD;
#pragma unroll
        for (int kk = 0; kk < 2; ++kk) {
          const __bf16* pk = kp + kk * 32 + half * 8;
          v16bf b = pack16(pk, pk + 16);
          s[t] = __builtin_amdgcn_wmma_f32_16x16x32_bf16(false, aq[kk], false, b, (short)0, s[t],
                                                         false, false);
        }
      }
      // ---- scale + decomposed rel-pos bias (all operands in registers) ----
#pragma unroll
      for (int r = 0; r < 8; ++r)
#pragma unroll
        for (int t = 0; t < 2; ++t)
          s[t][r] = s[t][r] * scale + bh[r] + bwreg[p][t][r];

      // ---- online softmax (rows span 16 lanes of a half-wave in C/D layout) ----
#pragma unroll
      for (int r = 0; r < 8; ++r) {
        float v = fmaxf(s[0][r], s[1][r]);
        v = fmaxf(v, __shfl_xor(v, 1, 16));
        v = fmaxf(v, __shfl_xor(v, 2, 16));
        v = fmaxf(v, __shfl_xor(v, 4, 16));
        v = fmaxf(v, __shfl_xor(v, 8, 16));
        const float nm   = fmaxf(mrow[r], v);
        const float corr = __expf(mrow[r] - nm);
        mrow[r] = nm;
        const float p0 = __expf(s[0][r] - nm);
        const float p1 = __expf(s[1][r] - nm);
        s[0][r] = p0; s[1][r] = p1;
        float rs = p0 + p1;
        rs += __shfl_xor(rs, 1, 16);
        rs += __shfl_xor(rs, 2, 16);
        rs += __shfl_xor(rs, 4, 16);
        rs += __shfl_xor(rs, 8, 16);
        lrow[r] = lrow[r] * corr + rs;
#pragma unroll
        for (int t = 0; t < 4; ++t) o[t][r] *= corr;
      }
      // ---- P: C-layout -> A-layout via LDS (16 x 32 bf16) ----
      __bf16* pb = pbuf[wid];
#pragma unroll
      for (int t = 0; t < 2; ++t)
#pragma unroll
        for (int r = 0; r < 8; ++r)
          pb[(r + half * 8) * 32 + t * 16 + lm] = f2bf(s[t][r]);
#if __has_builtin(__builtin_amdgcn_s_wait_dscnt)
      __builtin_amdgcn_s_wait_dscnt(0);
#endif
      const __bf16* pr = pb + lm * 32 + half * 8;
      v16bf pa = pack16(pr, pr + 16);
      // ---- O += P V  (V via pre-transposed vT: contiguous keys per lane) ----
#pragma unroll
      for (int t2 = 0; t2 < 4; ++t2) {
        const __bf16* vp = vT + ((size_t)(head * HD + t2 * 16 + lm)) * NTOK + k0 + half * 8;
        v16bf b = pack16(vp, vp + 16);
        o[t2] = __builtin_amdgcn_wmma_f32_16x16x32_bf16(false, pa, false, b, (short)0, o[t2],
                                                        false, false);
      }
    }
  }
  // ---- normalize and store (layout: [token][head*64 + d]) ----
#pragma unroll
  for (int r = 0; r < 8; ++r) {
    const int mr = m0 + r + half * 8;
    const float inv = 1.0f / lrow[r];
#pragma unroll
    for (int t2 = 0; t2 < 4; ++t2)
      outb[(size_t)mr * CDIM + head * HD + t2 * 16 + lm] = f2bf(o[t2][r] * inv);
  }
}

// ====================================================================
extern "C" void kernel_launch(void* const* d_in, const int* in_sizes, int n_in,
                              void* d_out, int out_size, void* d_ws, size_t ws_size,
                              hipStream_t stream) {
  const float* x         = (const float*)d_in[0];
  const float* qkv_w     = (const float*)d_in[1];
  const float* qkv_b     = (const float*)d_in[2];
  const float* proj_w    = (const float*)d_in[3];
  const float* proj_b    = (const float*)d_in[4];
  const float* rel_pos_h = (const float*)d_in[5];
  const float* rel_pos_w = (const float*)d_in[6];
  float* out = (float*)d_out;

  char* ws = (char*)d_ws;
  size_t off = 0;
  auto alloc = [&](size_t bytes) -> void* {
    void* p = ws + off;
    off += (bytes + 255) & ~(size_t)255;
    return p;
  };
  __bf16* xb       = (__bf16*)alloc((size_t)NTOK * CDIM * 2);
  __bf16* wqkv_t   = (__bf16*)alloc((size_t)QKVN * CDIM * 2);
  __bf16* wproj_t  = (__bf16*)alloc((size_t)CDIM * CDIM * 2);
  __bf16* qkv_buf  = (__bf16*)alloc((size_t)NTOK * QKVN * 2);
  __bf16* vTb      = (__bf16*)alloc((size_t)NHEAD * HD * NTOK * 2);
  float*  relh     = (float*) alloc((size_t)NHEAD * NTOK * HD * 4);
  float*  relw     = (float*) alloc((size_t)NHEAD * NTOK * HD * 4);
  __bf16* attn_out = (__bf16*)alloc((size_t)NTOK * CDIM * 2);

  // 1. precision conversions / transposes
  f32_to_bf16_kernel<<<(NTOK * CDIM) / 256, 256, 0, stream>>>(x, xb, NTOK * CDIM);
  transpose_f32_to_bf16_kernel<<<(CDIM * QKVN) / 256, 256, 0, stream>>>(qkv_w, wqkv_t, CDIM, QKVN);
  transpose_f32_to_bf16_kernel<<<(CDIM * CDIM) / 256, 256, 0, stream>>>(proj_w, wproj_t, CDIM, CDIM);

  // 2. qkv projection (4096x768 @ 768x2304 + b), bf16 out
  {
    int waves = (NTOK / 16) * (QKVN / 64);
    gemm_bf16_kernel<false><<<waves / 8, 256, 0, stream>>>(xb, wqkv_t, qkv_b, qkv_buf, nullptr,
                                                           NTOK, QKVN, CDIM);
  }

  // 3. V transpose + rel-pos bias tables
  transpose_v_kernel<<<(NHEAD * HD * NTOK) / 256, 256, 0, stream>>>(qkv_buf, vTb);
  rel_bias_kernel<<<(NHEAD * NTOK * HD) / 256, 256, 0, stream>>>(qkv_buf, rel_pos_h, relh, 0);
  rel_bias_kernel<<<(NHEAD * NTOK * HD) / 256, 256, 0, stream>>>(qkv_buf, rel_pos_w, relw, 1);

  // 4. flash attention with decomposed bias
  {
    int waves = NHEAD * (NTOK / 16);          // 3072
    flash_attn_kernel<<<waves / 8, 256, 0, stream>>>(qkv_buf, vTb, relh, relw, attn_out);
  }

  // 5. output projection (4096x768 @ 768x768 + b), fp32 out
  {
    int waves = (NTOK / 16) * (CDIM / 64);
    gemm_bf16_kernel<true><<<waves / 8, 256, 0, stream>>>(attn_out, wproj_t, proj_b, nullptr, out,
                                                          NTOK, CDIM, CDIM);
  }
}
